// MultiHeadAttention_29197187678242
// MI455X (gfx1250) — matrix operand order repaired and checked by reading, not compiled, not run on hardware
//
#include <hip/hip_runtime.h>
#include <hip/hip_bf16.h>

// ---------------------------------------------------------------------------
// MHA forward for MI455X (gfx1250, wave32). All matmuls via
// v_wmma_f32_16x16x32_bf16 (f32 accumulate). Pipeline:
//   1) f32->bf16 convert (q,k,v), f32->bf16 transpose-convert (Wq,Wk,Wv,Wo)
//   2) QKV projections: block-tiled WMMA GEMM, B panel staged to LDS with
//      double-buffered global_load_async_to_lds_b128 (ASYNCcnt)
//   3) flash attention (one wave per 16-row query tile, 32-key blocks)
//   4) output projection (same GEMM) -> f32 d_out
// ---------------------------------------------------------------------------

typedef __bf16 bf16;
typedef __attribute__((ext_vector_type(16))) __bf16 v16bf;
typedef __attribute__((ext_vector_type(8)))  __bf16 v8bf;
typedef __attribute__((ext_vector_type(8)))  float  v8f;

union V16 { v16bf v; v8bf h[2]; };

#define BB 4
#define TT 2048
#define DD 1024
#define HH 16
#define HD 64
#define MM (BB * TT)   // 8192
#define BM 128         // block M tile (8 waves x 16 rows)
#define BN 64          // block N tile (4 WMMA subtiles per wave)

__device__ inline v8f zero8() {
    v8f z;
    #pragma unroll
    for (int i = 0; i < 8; ++i) z[i] = 0.0f;
    return z;
}

__device__ inline float rmax16(float v) {
    #pragma unroll
    for (int off = 1; off < 16; off <<= 1)
        v = fmaxf(v, __shfl_xor(v, off, 32));
    return v;
}

__device__ inline float rsum16(float v) {
    #pragma unroll
    for (int off = 1; off < 16; off <<= 1)
        v += __shfl_xor(v, off, 32);
    return v;
}

// ---------------------------------------------------------------------------
// 1a) elementwise f32 -> bf16
// ---------------------------------------------------------------------------
__global__ void cvt_f32_to_bf16(const float* __restrict__ in,
                                bf16* __restrict__ out, int n) {
    int i = blockIdx.x * 256 + threadIdx.x;
    if (i < n) out[i] = (bf16)in[i];
}

// 1b) transpose-convert weight f32 [K,N] -> bf16 [N,K]
__global__ void cvtT_f32_to_bf16(const float* __restrict__ W,
                                 bf16* __restrict__ WT) {
    int i = blockIdx.x * 256 + threadIdx.x;   // output index, N-major
    int n = i >> 10;          // 0..1023
    int k = i & 1023;
    WT[i] = (bf16)W[k * DD + n];
}

// ---------------------------------------------------------------------------
// 2/4) GEMM: out[M,N] = A[M,K] * WT[N,K]^T + bias  (M=8192, N=K=1024)
// Block (256 thr = 8 waves) computes a 128x64 tile. Each wave owns a 16-row
// strip and 4 accumulators (16x64), reusing its A register tile across 4
// WMMAs. The shared B panel (64x32 bf16 per K-step) is staged into LDS with
// double-buffered async loads (global_load_async_to_lds_b128, ASYNCcnt).
// mode 0: f32 row-major (final projection)
// mode 1: bf16 head-major [B,H,T,HD] (Q,K)
// mode 2: bf16 transposed head-major [B,H,HD,T] (V)
// ---------------------------------------------------------------------------
__global__ void __launch_bounds__(256)
gemm_bf16_wmma(const bf16* __restrict__ A, const bf16* __restrict__ WT,
               const float* __restrict__ bias, float* __restrict__ out_f32,
               bf16* __restrict__ out_bf16, int mode) {
    __shared__ bf16 Bl[2][BN][32];      // 2 x 64 x 32 bf16 = 8 KB

    const int tid  = threadIdx.x;
    const int lane = tid & 31;
    const int wave = tid >> 5;
    const int half = lane >> 4;
    const int l16  = lane & 15;

    const int ncol0  = blockIdx.x * BN;          // 0..960
    const int mtile0 = blockIdx.y * BM;          // 0..8064
    const int m_row  = mtile0 + wave * 16 + l16;
    const bf16* arow = A + (size_t)m_row * DD;

    // Async B staging: thread t copies one 16-byte chunk per K-step.
    // n_local = t>>2 (0..63), chunk = t&3 covers k-range [chunk*8, chunk*8+8).
    const int n_local = tid >> 2;
    const int chunk   = tid & 3;
    const bf16* bsrc  = WT + (size_t)(ncol0 + n_local) * DD + chunk * 8;

    auto stage = [&](int buf, int kk) {
        unsigned lds_off = (unsigned)(uintptr_t)(&Bl[buf][n_local][chunk * 8]);
        unsigned long long gaddr = (unsigned long long)(uintptr_t)(bsrc + kk);
        asm volatile("global_load_async_to_lds_b128 %0, %1, off"
                     :: "v"(lds_off), "v"(gaddr) : "memory");
    };

    stage(0, 0);

    v8f acc[4];
    #pragma unroll
    for (int s = 0; s < 4; ++s) acc[s] = zero8();

    for (int kk = 0; kk < DD; kk += 32) {
        const int buf = (kk >> 5) & 1;
        if (kk + 32 < DD) {
            stage(buf ^ 1, kk + 32);
            asm volatile("s_wait_asynccnt 0x1" ::: "memory");
        } else {
            asm volatile("s_wait_asynccnt 0x0" ::: "memory");
        }
        __syncthreads();   // publish staged B panel across waves

        __builtin_prefetch(arow + kk + 256, 0, 1);   // global_prefetch_b8 (A stream)

        // A 16x32 register tile: lanes 0-15 hold K {kk..+7, kk+16..+23};
        // lanes 16-31 hold {kk+8..+15, kk+24..+31}.
        V16 a;
        const int ka = kk + (half ? 8 : 0);
        a.h[0] = *(const v8bf*)(arow + ka);
        a.h[1] = *(const v8bf*)(arow + ka + 16);

        // 4 B subtiles from LDS; lanes 0-15 read k {0..7,16..23}, lanes 16-31 k {8..15,24..31}.
        #pragma unroll
        for (int s = 0; s < 4; ++s) {
            V16 b;
            const bf16* bp = &Bl[buf][s * 16 + l16][half ? 8 : 0];
            b.h[0] = *(const v8bf*)(bp);
            b.h[1] = *(const v8bf*)(bp + 16);
            acc[s] = __builtin_amdgcn_wmma_f32_16x16x32_bf16(
                false, a.v, false, b.v, (short)0, acc[s], false, false);
        }
        __syncthreads();   // all waves done reading before buf is re-staged
    }

    #pragma unroll
    for (int s = 0; s < 4; ++s) {
        const int col = ncol0 + s * 16 + l16;
        const float bv = bias[col];
        #pragma unroll
        for (int i = 0; i < 8; ++i) {
            const int row = mtile0 + wave * 16 + i + 8 * half;   // C layout row
            const float val = acc[s][i] + bv;
            if (mode == 0) {
                out_f32[(size_t)row * DD + col] = val;
            } else {
                const int bI = row >> 11;       // row / T
                const int t  = row & (TT - 1);
                const int h  = col >> 6;        // col / HD
                const int hd = col & (HD - 1);
                if (mode == 1) {   // [B,H,T,HD]
                    out_bf16[(((size_t)(bI * HH + h)) * TT + t) * HD + hd] = (bf16)val;
                } else {           // [B,H,HD,T]
                    out_bf16[(((size_t)(bI * HH + h)) * HD + hd) * TT + t] = (bf16)val;
                }
            }
        }
    }
}

// ---------------------------------------------------------------------------
// 3) Flash attention. One wave handles one (b,h,16-query) tile, streaming
// 32-key blocks: S = Q*K^T (2 WMMAs per 16-key subtile), online softmax
// (16-lane shuffle reductions), P transposed via per-wave LDS, then
// out += P*V (4 WMMAs over HD=64).
// ---------------------------------------------------------------------------
__global__ void __launch_bounds__(256)
attn_fwd_wmma(const bf16* __restrict__ Qh,   // [B,H,T,HD]
              const bf16* __restrict__ Kh,   // [B,H,T,HD]
              const bf16* __restrict__ Vt,   // [B,H,HD,T]
              bf16* __restrict__ ctx) {      // [B*T, D] row-major
    __shared__ bf16 ldsP[8][16][32];

    const int lane = threadIdx.x & 31;
    const int wave = threadIdx.x >> 5;
    const int half = lane >> 4;
    const int l16  = lane & 15;
    const int c0   = half ? 8 : 0;

    const int h = blockIdx.y;
    const int b = blockIdx.z;
    const int qtile = blockIdx.x * 8 + wave;
    const int qbase = qtile * 16;

    const size_t bh = (size_t)(b * HH + h);
    const bf16* Qbase = Qh + bh * TT * HD;
    const bf16* Kbase = Kh + bh * TT * HD;
    const bf16* Vbase = Vt + bh * HD * TT;

    // Q tile in A layout (16x32 halves of HD=64).
    const bf16* qrow = Qbase + (size_t)(qbase + l16) * HD;
    V16 a_lo, a_hi;
    a_lo.h[0] = *(const v8bf*)(qrow + c0);
    a_lo.h[1] = *(const v8bf*)(qrow + c0 + 16);
    a_hi.h[0] = *(const v8bf*)(qrow + 32 + c0);
    a_hi.h[1] = *(const v8bf*)(qrow + 32 + c0 + 16);

    float m[8], l[8];
    v8f o[4];
    #pragma unroll
    for (int i = 0; i < 8; ++i) { m[i] = -1.0e30f; l[i] = 0.0f; }
    #pragma unroll
    for (int s = 0; s < 4; ++s) o[s] = zero8();

    const float scale = 0.03125f;     // 1/sqrt(D) = 1/32 (reference scaling)
    const int nkb = (qbase + 16 + 31) >> 5;

    for (int kb2 = 0; kb2 < nkb; ++kb2) {
        const int kbase = kb2 * 32;

        // K^T B tiles: column = key, K-dim = hd. Contiguous 16B loads.
        const int hb = half ? 8 : 0;
        const bf16* krow0 = Kbase + (size_t)(kbase + l16) * HD + hb;
        const bf16* krow1 = krow0 + 16 * HD;
        V16 b00, b01, b10, b11;
        b00.h[0] = *(const v8bf*)(krow0);      b00.h[1] = *(const v8bf*)(krow0 + 16);
        b01.h[0] = *(const v8bf*)(krow0 + 32); b01.h[1] = *(const v8bf*)(krow0 + 48);
        b10.h[0] = *(const v8bf*)(krow1);      b10.h[1] = *(const v8bf*)(krow1 + 16);
        b11.h[0] = *(const v8bf*)(krow1 + 32); b11.h[1] = *(const v8bf*)(krow1 + 48);

        v8f s0 = zero8(), s1 = zero8();
        s0 = __builtin_amdgcn_wmma_f32_16x16x32_bf16(false, a_lo.v, false, b00.v, (short)0, s0, false, false);
        s0 = __builtin_amdgcn_wmma_f32_16x16x32_bf16(false, a_hi.v, false, b01.v, (short)0, s0, false, false);
        s1 = __builtin_amdgcn_wmma_f32_16x16x32_bf16(false, a_lo.v, false, b10.v, (short)0, s1, false, false);
        s1 = __builtin_amdgcn_wmma_f32_16x16x32_bf16(false, a_hi.v, false, b11.v, (short)0, s1, false, false);

        const bool domask = (kbase + 31 > qbase);
        #pragma unroll
        for (int i = 0; i < 8; ++i) {
            s0[i] *= scale;
            s1[i] *= scale;
            if (domask) {
                const int row = qbase + i + 8 * half;
                const int key = kbase + l16;
                if (key > row)      s0[i] = -__builtin_inff();
                if (key + 16 > row) s1[i] = -__builtin_inff();
            }
        }

        // Online softmax with 16-lane row reductions.
        float p0[8], p1[8];
        #pragma unroll
        for (int i = 0; i < 8; ++i) {
            const float rm = rmax16(fmaxf(s0[i], s1[i]));
            const float mn = fmaxf(m[i], rm);
            const float sc = __expf(m[i] - mn);
            p0[i] = __expf(s0[i] - mn);
            p1[i] = __expf(s1[i] - mn);
            const float rs = rsum16(p0[i] + p1[i]);
            l[i] = l[i] * sc + rs;
            m[i] = mn;
            o[0][i] *= sc; o[1][i] *= sc; o[2][i] *= sc; o[3][i] *= sc;
        }

        // Transpose P (C layout -> A layout) via per-wave LDS.
        #pragma unroll
        for (int i = 0; i < 8; ++i) {
            const int r = i + 8 * half;
            ldsP[wave][r][l16]      = (bf16)p0[i];
            ldsP[wave][r][16 + l16] = (bf16)p1[i];
        }
        asm volatile("s_wait_dscnt 0x0" ::: "memory");  // wave-internal LDS RAW

        V16 pa;
        const bf16* prow = &ldsP[wave][l16][0];
        pa.h[0] = *(const v8bf*)(prow + c0);
        pa.h[1] = *(const v8bf*)(prow + c0 + 16);

        // PV: B tiles from transposed V, contiguous along keys.
        const int t0 = kbase + (half ? 8 : 0);
        #pragma unroll
        for (int s = 0; s < 4; ++s) {
            const int col = s * 16 + l16;                  // hd
            const bf16* vptr = Vbase + (size_t)col * TT + t0;
            V16 bvv;
            bvv.h[0] = *(const v8bf*)(vptr);
            bvv.h[1] = *(const v8bf*)(vptr + 16);
            o[s] = __builtin_amdgcn_wmma_f32_16x16x32_bf16(
                false, pa.v, false, bvv.v, (short)0, o[s], false, false);
        }
    }

    // Normalize and store context bf16 in [B*T, D] row-major.
    float inv[8];
    #pragma unroll
    for (int i = 0; i < 8; ++i) inv[i] = 1.0f / l[i];
    #pragma unroll
    for (int s = 0; s < 4; ++s) {
        #pragma unroll
        for (int i = 0; i < 8; ++i) {
            const int row = qbase + i + 8 * half;
            const int col = h * HD + s * 16 + l16;
            ctx[((size_t)(b * TT) + row) * DD + col] = (bf16)(o[s][i] * inv[i]);
        }
    }
}

// ---------------------------------------------------------------------------
// Launcher
// ---------------------------------------------------------------------------
extern "C" void kernel_launch(void* const* d_in, const int* in_sizes, int n_in,
                              void* d_out, int out_size, void* d_ws, size_t ws_size,
                              hipStream_t stream) {
    const float* q  = (const float*)d_in[0];
    const float* k  = (const float*)d_in[1];
    const float* v  = (const float*)d_in[2];
    const float* Wq = (const float*)d_in[3];
    const float* bq = (const float*)d_in[4];
    const float* Wk = (const float*)d_in[5];
    const float* bk = (const float*)d_in[6];
    const float* Wv = (const float*)d_in[7];
    const float* bv = (const float*)d_in[8];
    const float* Wo = (const float*)d_in[9];
    const float* bo = (const float*)d_in[10];
    float* out = (float*)d_out;

    char* ws = (char*)d_ws;
    const size_t ACT = (size_t)MM * DD * sizeof(bf16);   // 16 MB
    const size_t WGT = (size_t)DD * DD * sizeof(bf16);   //  2 MB
    bf16* qb   = (bf16*)(ws);
    bf16* kb   = (bf16*)(ws + ACT);
    bf16* vb   = (bf16*)(ws + 2 * ACT);
    bf16* WqT  = (bf16*)(ws + 3 * ACT);
    bf16* WkT  = (bf16*)(ws + 3 * ACT + WGT);
    bf16* WvT  = (bf16*)(ws + 3 * ACT + 2 * WGT);
    bf16* WoT  = (bf16*)(ws + 3 * ACT + 3 * WGT);
    bf16* Qh   = (bf16*)(ws + 3 * ACT + 4 * WGT);
    bf16* Kh   = (bf16*)(ws + 4 * ACT + 4 * WGT);
    bf16* Vt   = (bf16*)(ws + 5 * ACT + 4 * WGT);
    bf16* ctx  = (bf16*)(ws + 6 * ACT + 4 * WGT);

    const int nAct = MM * DD;                  // 8 M elements
    // 1) precision conversion
    cvt_f32_to_bf16<<<nAct / 256, 256, 0, stream>>>(q, qb, nAct);
    cvt_f32_to_bf16<<<nAct / 256, 256, 0, stream>>>(k, kb, nAct);
    cvt_f32_to_bf16<<<nAct / 256, 256, 0, stream>>>(v, vb, nAct);
    cvtT_f32_to_bf16<<<(DD * DD) / 256, 256, 0, stream>>>(Wq, WqT);
    cvtT_f32_to_bf16<<<(DD * DD) / 256, 256, 0, stream>>>(Wk, WkT);
    cvtT_f32_to_bf16<<<(DD * DD) / 256, 256, 0, stream>>>(Wv, WvT);
    cvtT_f32_to_bf16<<<(DD * DD) / 256, 256, 0, stream>>>(Wo, WoT);

    // 2) QKV projections
    dim3 gGemm(DD / BN, MM / BM);              // (16, 64)
    gemm_bf16_wmma<<<gGemm, 256, 0, stream>>>(qb, WqT, bq, nullptr, Qh, 1);
    gemm_bf16_wmma<<<gGemm, 256, 0, stream>>>(kb, WkT, bk, nullptr, Kh, 1);
    gemm_bf16_wmma<<<gGemm, 256, 0, stream>>>(vb, WvT, bv, nullptr, Vt, 2);

    // 3) attention
    dim3 gAttn(TT / 16 / 8, HH, BB);           // (16, 16, 4)
    attn_fwd_wmma<<<gAttn, 256, 0, stream>>>(Qh, Kh, Vt, ctx);

    // 4) output projection -> f32
    gemm_bf16_wmma<<<gGemm, 256, 0, stream>>>(ctx, WoT, bo, out, nullptr, 0);
}
